// SetAbstractMSG_29076928594034
// MI455X (gfx1250) — matrix operand order repaired and checked by reading, not compiled, not run on hardware
//
#include <hip/hip_runtime.h>

typedef __attribute__((ext_vector_type(16))) _Float16 v16h;
typedef __attribute__((ext_vector_type(8)))  _Float16 v8h;
typedef __attribute__((ext_vector_type(4)))  _Float16 v4h;
typedef __attribute__((ext_vector_type(8)))  float    v8f;

#define B_      16
#define N_      4096
#define S_      1024
#define CIN_    64
#define CENTERS (B_ * S_)
#define CMAX    128
#define OUTC    320
#define STATS_WAVES 4096

// ---------------------------------------------------------------------------
// Farthest point sampling: one block per batch, xyz staged in LDS,
// distances in registers (4 points/thread), LDS tree argmax.
// ---------------------------------------------------------------------------
__global__ __launch_bounds__(1024) void fps_kernel(const float* __restrict__ xyz,
                                                   float* __restrict__ out_fps,
                                                   float* __restrict__ fps_xyz)
{
  __shared__ float sx[N_ * 3];
  __shared__ float rv[1024];
  __shared__ int   ri[1024];
  __shared__ int   sFar;
  const int b = blockIdx.x, t = threadIdx.x;
  for (int i = t; i < N_ * 3; i += 1024) sx[i] = xyz[(size_t)b * N_ * 3 + i];
  float dist[4];
#pragma unroll
  for (int i = 0; i < 4; ++i) dist[i] = 1e10f;
  if (t == 0) sFar = 0;
  __syncthreads();
  for (int it = 0; it < S_; ++it) {
    const int far = sFar;
    const float cx = sx[far * 3 + 0], cy = sx[far * 3 + 1], cz = sx[far * 3 + 2];
    float best = -1.0f;
    int bi = t * 4;
#pragma unroll
    for (int i = 0; i < 4; ++i) {
      const int p = t * 4 + i;
      const float dx = sx[p * 3 + 0] - cx, dy = sx[p * 3 + 1] - cy, dz = sx[p * 3 + 2] - cz;
      const float d = dx * dx + dy * dy + dz * dz;
      dist[i] = fminf(dist[i], d);
      if (dist[i] > best) { best = dist[i]; bi = p; }
    }
    rv[t] = best; ri[t] = bi;
    __syncthreads();
    for (int sft = 512; sft > 0; sft >>= 1) {
      if (t < sft) {
        const float ov = rv[t + sft]; const int oi = ri[t + sft];
        if (ov > rv[t] || (ov == rv[t] && oi < ri[t])) { rv[t] = ov; ri[t] = oi; }
      }
      __syncthreads();
    }
    if (t == 0) {
      const size_t o = ((size_t)b * S_ + it) * 3;
      out_fps[o + 0] = cx; out_fps[o + 1] = cy; out_fps[o + 2] = cz;
      fps_xyz[o + 0] = cx; fps_xyz[o + 1] = cy; fps_xyz[o + 2] = cz;
      sFar = ri[0];
    }
    __syncthreads();
  }
}

// ---------------------------------------------------------------------------
// Ball grouping: one wave per center; ballot+prefix-popcount appends the
// in-radius indices in ascending order, padded with the first hit.
// ---------------------------------------------------------------------------
__global__ __launch_bounds__(128) void ball_kernel(const float* __restrict__ xyz,
                                                   const float* __restrict__ fps_xyz,
                                                   int* __restrict__ gidx,
                                                   float r2, int nsample)
{
  const int wid  = (int)((blockIdx.x * blockDim.x + threadIdx.x) >> 5);
  const int lane = threadIdx.x & 31;
  if (wid >= CENTERS) return;
  const int b = wid >> 10;
  const float cx = fps_xyz[wid * 3 + 0], cy = fps_xyz[wid * 3 + 1], cz = fps_xyz[wid * 3 + 2];
  int* out = gidx + (size_t)wid * nsample;
  int cnt = 0;
  for (int base = 0; base < N_ && cnt < nsample; base += 32) {
    const int j = base + lane;
    const float* p = xyz + ((size_t)b * N_ + j) * 3;
    const float dx = p[0] - cx, dy = p[1] - cy, dz = p[2] - cz;
    const bool within = (dx * dx + dy * dy + dz * dz) <= r2;
    const unsigned mask = (unsigned)__ballot(within);
    const int pos = cnt + __popc(mask & ((1u << lane) - 1u));
    if (within && pos < nsample) out[pos] = j;
    cnt += __popc(mask);
  }
  if (cnt > nsample) cnt = nsample;
  const int first = (cnt > 0) ? out[0] : (N_ - 1);  // empty ball -> clamp like JAX gather
  for (int p = cnt + lane; p < nsample; p += 32) out[p] = first;
}

// ---------------------------------------------------------------------------
// Weight prep / BN fold (f32 -> padded f16)
// ---------------------------------------------------------------------------
__global__ void prep_w_kernel(const float* __restrict__ W, int Cout, int cinReal, int cinPad,
                              _Float16* __restrict__ Wf)
{
  const int total = Cout * cinPad;
  for (int i = blockIdx.x * blockDim.x + threadIdx.x; i < total; i += gridDim.x * blockDim.x) {
    const int o = i / cinPad, c = i % cinPad;
    Wf[i] = (c < cinReal) ? (_Float16)W[(size_t)o * cinReal + c] : (_Float16)0.f;
  }
}

__global__ void fold_w_kernel(const float* __restrict__ W, const float* __restrict__ bias,
                              const float* __restrict__ gamma, const float* __restrict__ beta,
                              const float* __restrict__ stats, float minv,
                              int Cout, int cinReal, int cinPad,
                              _Float16* __restrict__ Wf, float* __restrict__ bf)
{
  const int total = Cout * cinPad;
  for (int i = blockIdx.x * blockDim.x + threadIdx.x; i < total; i += gridDim.x * blockDim.x) {
    const int o = i / cinPad, c = i % cinPad;
    const float mean = stats[o * 2] * minv;
    const float var  = stats[o * 2 + 1] * minv - mean * mean;
    const float g = gamma[o] * rsqrtf(var + 1e-5f);
    Wf[i] = (c < cinReal) ? (_Float16)(W[(size_t)o * cinReal + c] * g) : (_Float16)0.f;
    if (c == 0) bf[o] = (bias[o] - mean) * g + beta[o];
  }
}

// Deterministic second-stage reduction of per-wave partial (sum, sumsq).
__global__ __launch_bounds__(128) void stats_reduce_kernel(const float* __restrict__ partial,
                                                           float* __restrict__ stats, int nw)
{
  __shared__ float sa[128], sb[128];
  const int c = blockIdx.x, t = threadIdx.x;
  float a = 0.f, b = 0.f;
  for (int wv = t; wv < nw; wv += 128) {
    a += partial[((size_t)wv * CMAX + c) * 2 + 0];
    b += partial[((size_t)wv * CMAX + c) * 2 + 1];
  }
  sa[t] = a; sb[t] = b; __syncthreads();
  for (int s = 64; s > 0; s >>= 1) {
    if (t < s) { sa[t] += sa[t + s]; sb[t] += sb[t + s]; }
    __syncthreads();
  }
  if (t == 0) { stats[c * 2] = sa[0]; stats[c * 2 + 1] = sb[0]; }
}

// ---------------------------------------------------------------------------
// A fragment (weights): contiguous k order, k = 16*(lane/16) + s; two aligned v8h loads.
// ---------------------------------------------------------------------------
template <int CIN>
__device__ __forceinline__ v16h load_a_frag(const _Float16* __restrict__ w, int mo, int kb, int lane)
{
  const int M  = mo * 16 + (lane & 15);
  const int hi = (lane & 16) ? 16 : 0;
  const _Float16* row = w + M * CIN + kb * 32 + hi;
  const v8h lo = *(const v8h*)(row);
  const v8h hx = *(const v8h*)(row + 8);
  v16h a;
#pragma unroll
  for (int i = 0; i < 8; ++i) { a[i] = lo[i]; a[8 + i] = hx[i]; }
  return a;
}

// Activation tiles live in LDS in B-fragment layout:
//   element (k, n) -> half index (k>>5)*512 + ((k & 16) + n)*16 + (k & 15)
// so a B fragment for k-block kb is 16 contiguous halves per lane.
__device__ __forceinline__ int frag_addr(int k, int n)
{
  return ((k >> 5) << 9) + (((k & 16) + n) << 4) + (k & 15);
}

// ---------------------------------------------------------------------------
// One MLP layer on a 16-column tile, fully specialized.
// ACT: 0 = ReLU + store to xn (fragment layout), 1 = stats accumulate (raw),
//      2 = ReLU + running max.
// ---------------------------------------------------------------------------
template <int CIN, int COUT, int ACT>
__device__ __forceinline__ void run_layer(const _Float16* __restrict__ W,
                                          const float* __restrict__ bias,  // LDS
                                          const _Float16* __restrict__ xc,
                                          _Float16* __restrict__ xn,
                                          int lane, int hi8, int n,
                                          float (*s1)[8], float (*sq)[8], float (*vmax)[8])
{
  constexpr int KB = CIN / 32;
  constexpr int RT = COUT / 16;
  v16h bf[KB];
#pragma unroll
  for (int kb = 0; kb < KB; ++kb)
    bf[kb] = *(const v16h*)(xc + kb * 512 + lane * 16);
#pragma unroll
  for (int mo = 0; mo < RT; ++mo) {
    v8f acc = {0.f, 0.f, 0.f, 0.f, 0.f, 0.f, 0.f, 0.f};
#pragma unroll
    for (int kb = 0; kb < KB; ++kb) {
      const v16h a = load_a_frag<CIN>(W, mo, kb, lane);
      acc = __builtin_amdgcn_wmma_f32_16x16x32_f16(false, a, false, bf[kb], (short)0, acc, false, false);
    }
    float vals[8];
#pragma unroll
    for (int r = 0; r < 8; ++r) {
      float v = acc[r] + bias[mo * 16 + hi8 + r];
      if (ACT != 1) v = fmaxf(v, 0.f);
      vals[r] = v;
    }
    if constexpr (ACT == 0) {
      v8h pk;
#pragma unroll
      for (int r = 0; r < 8; ++r) pk[r] = (_Float16)vals[r];
      *(v8h*)(xn + (mo >> 1) * 512 + (((mo & 1) * 16) + n) * 16 + hi8) = pk;
    } else if constexpr (ACT == 1) {
#pragma unroll
      for (int r = 0; r < 8; ++r) { s1[mo][r] += vals[r]; sq[mo][r] += vals[r] * vals[r]; }
    } else {
#pragma unroll
      for (int r = 0; r < 8; ++r) vmax[mo][r] = fmaxf(vmax[mo][r], vals[r]);
    }
  }
}

// ---------------------------------------------------------------------------
// Fused MLP pass, fully specialized on (MODE, channels, K, L).
// MODE==0: layers 1..L-1 folded+ReLU, layer L raw -> per-wave (sum,sumsq).
// MODE==1: all 3 folded+ReLU, max over K neighbors -> output.
// One wave per center; per-wave LDS activation tiles (DS ops are in-order
// within a wave, so no barriers are needed after the initial weight staging).
// ---------------------------------------------------------------------------
template <int MODE, int C1, int C2, int C3, int KS, int L>
__global__ __launch_bounds__(64) void mlp_pass_kernel(
    const int* __restrict__ gidx, const float* __restrict__ pfeat,
    const float* __restrict__ xyz, const float* __restrict__ fps_xyz,
    const _Float16* __restrict__ W1, const float* __restrict__ B1g,
    const _Float16* __restrict__ W2, const float* __restrict__ B2g,
    const _Float16* __restrict__ W3, const float* __restrict__ B3g,
    float* __restrict__ partial, float* __restrict__ outFeat, int outOff)
{
  constexpr int CP1 = 96, CP2 = C1, CP3 = C2;
  constexpr int CL  = (L == 1) ? C1 : ((L == 2) ? C2 : C3);
  constexpr int RTL = CL / 16;
  constexpr int AL  = (MODE == 0) ? 1 : 2;   // action for the last executed layer

  __shared__ alignas(64) _Float16 sW1[C1 * CP1];
  __shared__ alignas(64) _Float16 sW2[(L >= 2) ? C2 * CP2 : 1];
  __shared__ alignas(64) _Float16 sW3[(L >= 3) ? C3 * CP3 : 1];
  __shared__ alignas(64) _Float16 sX[2][2][96 * 16];
  __shared__ float sB1[C1];
  __shared__ float sB2[(L >= 2) ? C2 : 1];
  __shared__ float sB3[(L >= 3) ? C3 : 1];

  const int t = threadIdx.x;
  const int w = t >> 5;
  const int lane = t & 31;
  for (int i = t; i < C1 * CP1; i += 64) sW1[i] = W1[i];
  for (int i = t; i < C1; i += 64) sB1[i] = B1g[i];
  if constexpr (L >= 2) {
    for (int i = t; i < C2 * CP2; i += 64) sW2[i] = W2[i];
    for (int i = t; i < C2; i += 64) sB2[i] = B2g[i];
  }
  if constexpr (L >= 3) {
    for (int i = t; i < C3 * CP3; i += 64) sW3[i] = W3[i];
    for (int i = t; i < C3; i += 64) sB3[i] = B3g[i];
  }
  __syncthreads();

  const int hi8 = (lane & 16) ? 8 : 0;
  const int n   = lane & 15;
  const int waveId     = blockIdx.x * 2 + w;
  const int totalWaves = gridDim.x * 2;

  float s1[RTL][8], sq[RTL][8];
#pragma unroll
  for (int a = 0; a < RTL; ++a)
#pragma unroll
    for (int r = 0; r < 8; ++r) { s1[a][r] = 0.f; sq[a][r] = 0.f; }

  for (int center = waveId; center < CENTERS; center += totalWaves) {
    const int b = center >> 10;
    float vmax[RTL][8];
    if constexpr (MODE == 1) {
#pragma unroll
      for (int a = 0; a < RTL; ++a)
#pragma unroll
        for (int r = 0; r < 8; ++r) vmax[a][r] = -3.0e38f;
    }
#pragma unroll 1
    for (int tile = 0; tile < KS / 16; ++tile) {
      _Float16* x0 = sX[w][0];
      _Float16* x1 = sX[w][1];
      { // gather 16 columns into fragment layout: 64 features + 3 rel-xyz + pad
        const int ncol = lane >> 1, hsel = lane & 1;
        const int j = gidx[(size_t)center * KS + tile * 16 + ncol];
        // zero the k-block 2 region (k = 64..95): 16 contiguous halves per lane
        v16h z;
#pragma unroll
        for (int i = 0; i < 16; ++i) z[i] = (_Float16)0.f;
        *(v16h*)(x0 + 1024 + lane * 16) = z;
        const float4* frow = (const float4*)(pfeat + ((size_t)b * N_ + j) * CIN_ + hsel * 32);
#pragma unroll
        for (int q = 0; q < 8; ++q) {
          const float4 f = frow[q];
          const int c0 = hsel * 32 + q * 4;
          v4h pk;
          pk[0] = (_Float16)f.x; pk[1] = (_Float16)f.y;
          pk[2] = (_Float16)f.z; pk[3] = (_Float16)f.w;
          *(v4h*)(x0 + frag_addr(c0, ncol)) = pk;
        }
        if (hsel == 0) {
          const float* p = xyz + ((size_t)b * N_ + j) * 3;
          x0[frag_addr(64, ncol)] = (_Float16)(p[0] - fps_xyz[(size_t)center * 3 + 0]);
          x0[frag_addr(65, ncol)] = (_Float16)(p[1] - fps_xyz[(size_t)center * 3 + 1]);
          x0[frag_addr(66, ncol)] = (_Float16)(p[2] - fps_xyz[(size_t)center * 3 + 2]);
        }
      }
      if constexpr (L == 1) {
        run_layer<CP1, C1, AL>(sW1, sB1, x0, x1, lane, hi8, n, s1, sq, vmax);
      } else {
        run_layer<CP1, C1, 0>(sW1, sB1, x0, x1, lane, hi8, n, s1, sq, vmax);
        if constexpr (L == 2) {
          run_layer<CP2, C2, AL>(sW2, sB2, x1, x0, lane, hi8, n, s1, sq, vmax);
        } else {
          run_layer<CP2, C2, 0>(sW2, sB2, x1, x0, lane, hi8, n, s1, sq, vmax);
          run_layer<CP3, C3, AL>(sW3, sB3, x0, x1, lane, hi8, n, s1, sq, vmax);
        }
      }
    }
    if constexpr (MODE == 1) {
#pragma unroll
      for (int mo = 0; mo < RTL; ++mo)
#pragma unroll
        for (int r = 0; r < 8; ++r) {
          float m = vmax[mo][r];
          m = fmaxf(m, __shfl_xor(m, 1, 32));
          m = fmaxf(m, __shfl_xor(m, 2, 32));
          m = fmaxf(m, __shfl_xor(m, 4, 32));
          m = fmaxf(m, __shfl_xor(m, 8, 32));
          if (n == 0) {
            const int ch = mo * 16 + hi8 + r;
            outFeat[(size_t)center * OUTC + outOff + ch] = m;
          }
        }
    }
  }
  if constexpr (MODE == 0) {
#pragma unroll
    for (int mo = 0; mo < RTL; ++mo)
#pragma unroll
      for (int r = 0; r < 8; ++r) {
        float a = s1[mo][r], bq = sq[mo][r];
        a  += __shfl_xor(a, 1, 32);  a  += __shfl_xor(a, 2, 32);
        a  += __shfl_xor(a, 4, 32);  a  += __shfl_xor(a, 8, 32);
        bq += __shfl_xor(bq, 1, 32); bq += __shfl_xor(bq, 2, 32);
        bq += __shfl_xor(bq, 4, 32); bq += __shfl_xor(bq, 8, 32);
        if (n == 0) {
          const int ch = mo * 16 + hi8 + r;
          partial[((size_t)waveId * CMAX + ch) * 2 + 0] = a;
          partial[((size_t)waveId * CMAX + ch) * 2 + 1] = bq;
        }
      }
  }
}

// ---------------------------------------------------------------------------
// Host-side per-scale driver (fully specialized kernels).
// prm[l*4 + {0,1,2,3}] = W, bias, gamma, beta of layer l for this scale.
// ---------------------------------------------------------------------------
template <int C1, int C2, int C3, int KS>
static void run_scale(const float* xyz, const float* pfeat, float* fps_xyz, int* gidx,
                      _Float16* const* wraw, _Float16* const* wfold, float* bfold,
                      float* stats, float* partial, float* out_feat,
                      const float* const* prm, float r2, int outOff, hipStream_t stream)
{
  const int cinReal[3] = {67, C1, C2};
  const int cinPad[3]  = {96, C1, C2};
  const int Cs[3]      = {C1, C2, C3};

  ball_kernel<<<CENTERS / 4, 128, 0, stream>>>(xyz, fps_xyz, gidx, r2, KS);
  for (int l = 0; l < 3; ++l)
    prep_w_kernel<<<48, 256, 0, stream>>>(prm[l * 4 + 0], Cs[l], cinReal[l], cinPad[l], wraw[l]);

  const float minv = 1.0f / ((float)CENTERS * (float)KS);

  // layer 1 stats -> fold
  mlp_pass_kernel<0, C1, C2, C3, KS, 1><<<STATS_WAVES / 2, 64, 0, stream>>>(
      gidx, pfeat, xyz, fps_xyz, wraw[0], prm[1], wraw[1], prm[5], wraw[2], prm[9],
      partial, out_feat, outOff);
  stats_reduce_kernel<<<C1, 128, 0, stream>>>(partial, stats, STATS_WAVES);
  fold_w_kernel<<<48, 256, 0, stream>>>(prm[0], prm[1], prm[2], prm[3], stats, minv,
                                        C1, 67, 96, wfold[0], bfold + 0);
  // layer 2 stats -> fold
  mlp_pass_kernel<0, C1, C2, C3, KS, 2><<<STATS_WAVES / 2, 64, 0, stream>>>(
      gidx, pfeat, xyz, fps_xyz, wfold[0], bfold + 0, wraw[1], prm[5], wraw[2], prm[9],
      partial, out_feat, outOff);
  stats_reduce_kernel<<<C2, 128, 0, stream>>>(partial, stats, STATS_WAVES);
  fold_w_kernel<<<48, 256, 0, stream>>>(prm[4], prm[5], prm[6], prm[7], stats, minv,
                                        C2, C1, C1, wfold[1], bfold + CMAX);
  // layer 3 stats -> fold
  mlp_pass_kernel<0, C1, C2, C3, KS, 3><<<STATS_WAVES / 2, 64, 0, stream>>>(
      gidx, pfeat, xyz, fps_xyz, wfold[0], bfold + 0, wfold[1], bfold + CMAX, wraw[2], prm[9],
      partial, out_feat, outOff);
  stats_reduce_kernel<<<C3, 128, 0, stream>>>(partial, stats, STATS_WAVES);
  fold_w_kernel<<<48, 256, 0, stream>>>(prm[8], prm[9], prm[10], prm[11], stats, minv,
                                        C3, C2, C2, wfold[2], bfold + 2 * CMAX);
  // final fused pass: all folded, ReLU, max over K, write out
  mlp_pass_kernel<1, C1, C2, C3, KS, 3><<<CENTERS / 2, 64, 0, stream>>>(
      gidx, pfeat, xyz, fps_xyz, wfold[0], bfold + 0, wfold[1], bfold + CMAX,
      wfold[2], bfold + 2 * CMAX, partial, out_feat, outOff);
}

// ---------------------------------------------------------------------------
extern "C" void kernel_launch(void* const* d_in, const int* in_sizes, int n_in,
                              void* d_out, int out_size, void* d_ws, size_t ws_size,
                              hipStream_t stream)
{
  const float* xyz   = (const float*)d_in[0];
  const float* pfeat = (const float*)d_in[1];
  float* out_fps  = (float*)d_out;
  float* out_feat = (float*)d_out + (size_t)CENTERS * 3;

  char* ws = (char*)d_ws;
  size_t off = 0;
  float* fps_xyz = (float*)(ws + off); off += (size_t)CENTERS * 3 * 4;
  off = (off + 255) & ~(size_t)255;
  int* gidx = (int*)(ws + off); off += (size_t)CENTERS * 128 * 4;
  off = (off + 255) & ~(size_t)255;
  _Float16* wraw[3]; _Float16* wfold[3];
  for (int l = 0; l < 3; ++l) { wraw[l]  = (_Float16*)(ws + off); off += 128 * 96 * 2; }
  for (int l = 0; l < 3; ++l) { wfold[l] = (_Float16*)(ws + off); off += 128 * 96 * 2; }
  float* bfold   = (float*)(ws + off); off += 3 * CMAX * 4;
  float* stats   = (float*)(ws + off); off += CMAX * 2 * 4;
  float* partial = (float*)(ws + off); off += (size_t)STATS_WAVES * CMAX * 2 * 4;

  fps_kernel<<<B_, 1024, 0, stream>>>(xyz, out_fps, fps_xyz);

  const float* prm0[12]; const float* prm1[12]; const float* prm2[12];
  for (int i = 0; i < 12; ++i) {
    prm0[i] = (const float*)d_in[2 + i];
    prm1[i] = (const float*)d_in[2 + 12 + i];
    prm2[i] = (const float*)d_in[2 + 24 + i];
  }

  run_scale<32, 32, 64, 16>(xyz, pfeat, fps_xyz, gidx, wraw, wfold, bfold, stats, partial,
                            out_feat, prm0, 0.01f, 0, stream);
  run_scale<64, 64, 128, 32>(xyz, pfeat, fps_xyz, gidx, wraw, wfold, bfold, stats, partial,
                             out_feat, prm1, 0.04f, 64, stream);
  run_scale<64, 96, 128, 128>(xyz, pfeat, fps_xyz, gidx, wraw, wfold, bfold, stats, partial,
                              out_feat, prm2, 0.16f, 192, stream);
  (void)in_sizes; (void)n_in; (void)out_size; (void)ws_size;
}
